// GenLSTM_33414845563043
// MI455X (gfx1250) — matrix operand order repaired and checked by reading, not compiled, not run on hardware
//
#include <hip/hip_runtime.h>

// ---------------------------------------------------------------------------
// GenLSTM for MI455X (gfx1250): batch-parallel recurrent LSTM, one wave per
// 16 batch rows, bf16 WMMA (v_wmma_f32_16x16x32_bf16) for the gate GEMM with
// the input matmul + biases folded into an augmented K=96 contraction.
// Activations use v_exp_f32 + v_rcp_f32 (no IEEE div expansion).
// ---------------------------------------------------------------------------

typedef __bf16 v16bf __attribute__((ext_vector_type(16)));
typedef float  v8f   __attribute__((ext_vector_type(8)));

constexpr int Bn  = 8192;
constexpr int Ln  = 512;
constexpr int ND  = 4;
constexpr int SD  = 4;
constexpr int Hn  = 64;
constexpr int NG  = 4 * Hn;        // 256 gate outputs (i,f,g,o)
constexpr int INn = SD + ND + 1;   // 9
constexpr int KA  = 96;            // augmented K: h(64) x(4) noise(4) dt(1) one(1) pad(22)
constexpr int WAVES = 4;
constexpr int TPB   = WAVES * 32;              // 128 threads
constexpr int BATCH_PER_WAVE  = 16;
constexpr int BATCH_PER_BLOCK = WAVES * BATCH_PER_WAVE; // 64

__device__ __forceinline__ void lds_fence() {
  // wave32 lockstep: all lanes' ds stores complete before subsequent ds loads
  asm volatile("s_wait_dscnt 0" ::: "memory");
}

__device__ __forceinline__ v8f wmma_bf16(v16bf a, v16bf b, v8f c) {
  return __builtin_amdgcn_wmma_f32_16x16x32_bf16(
      /*neg_a=*/false, a, /*neg_b=*/false, b,
      /*c_mod=*/(short)0, c, /*reuse_a=*/false, /*reuse_b=*/false);
}

// A-matrix fragment (16x32 bf16): lanes 0-15 (M=lane) hold K = kofs+0..7 and
// kofs+16..23; lanes 16-31 (M=lane-16) hold K = kofs+8..15 and kofs+24..31.
__device__ __forceinline__ v16bf ldA(const __bf16* abase, int lane, int ktile) {
  const int m    = lane & 15;
  const int kofs = ktile * 32 + ((lane >> 4) << 3);
  const __bf16* p = abase + m * KA + kofs;
  union { uint4 q[2]; v16bf v; } u;
  u.q[0] = *(const uint4*)(p);        // K = kofs .. kofs+7
  u.q[1] = *(const uint4*)(p + 16);   // K = kofs+16 .. kofs+23
  return u.v;
}

// B-matrix fragment (32x16 bf16), weights stored [N][K] row-major in LDS:
// lanes 0-15 (N=lane) hold K = kofs+0..7 and kofs+16..23; lanes 16-31 hold K = kofs+8..15 and kofs+24..31.
__device__ __forceinline__ v16bf ldB(const __bf16* bbase, int lane, int ktile,
                                     int rstride) {
  const int n    = lane & 15;
  const int kofs = ktile * 32 + ((lane >> 4) << 3);
  const __bf16* p = bbase + n * rstride + kofs;
  union { uint4 q[2]; v16bf v; } u;
  u.q[0] = *(const uint4*)(p);        // K = kofs .. kofs+7
  u.q[1] = *(const uint4*)(p + 16);   // K = kofs+16 .. kofs+23
  return u.v;
}

// v_rcp_f32: ~1 ulp, no div expansion; plenty for bf16-fed gate math
__device__ __forceinline__ float fast_rcp(float x) {
  return __builtin_amdgcn_rcpf(x);
}
__device__ __forceinline__ float fast_sigmoid(float x) {
  return fast_rcp(1.0f + __expf(-x));          // v_exp_f32 + v_rcp_f32
}
__device__ __forceinline__ float fast_tanh(float x) {
  const float e = __expf(2.0f * x);
  return 1.0f - 2.0f * fast_rcp(e + 1.0f);     // saturates correctly at +-1
}

__global__ __launch_bounds__(TPB)
void genlstm_wmma_kernel(const float* __restrict__ noise,
                         const float* __restrict__ t,
                         const float* __restrict__ W_ih,
                         const float* __restrict__ W_hh,
                         const float* __restrict__ b_ih,
                         const float* __restrict__ b_hh,
                         const float* __restrict__ Wo,
                         const float* __restrict__ bo,
                         float* __restrict__ out) {
  __shared__ __bf16 sW[NG * KA];           // augmented weights [n=256][k=96]
  __shared__ __bf16 sWo[16 * Hn];          // Wo^T padded to 16 rows [n=16][k=64]
  __shared__ __bf16 sA[WAVES][16 * KA];    // per-wave augmented A staging

  const int tid  = threadIdx.x;
  const int wave = tid >> 5;
  const int lane = tid & 31;
  const int nIdx = lane & 15;
  const int hi   = lane >> 4;

  // ---- stage weights into LDS (once) -----------------------------------
  for (int idx = tid; idx < NG * KA; idx += TPB) {
    const int n = idx / KA, k = idx - n * KA;
    float w;
    if (k < Hn)             w = W_hh[n * Hn + k];
    else if (k < Hn + INn)  w = W_ih[n * INn + (k - Hn)];   // inp = [x, noise, dt]
    else if (k == Hn + INn) w = b_ih[n] + b_hh[n];          // bias column (a=1)
    else                    w = 0.0f;
    sW[idx] = (__bf16)w;
  }
  for (int idx = tid; idx < 16 * Hn; idx += TPB) {
    const int n = idx / Hn, k = idx - n * Hn;
    sWo[idx] = (__bf16)((n < SD) ? Wo[n * Hn + k] : 0.0f);
  }
  __bf16* myA = sA[wave];
  for (int i = lane; i < 16 * KA; i += 32) myA[i] = (__bf16)0.0f;
  if (lane < 16) myA[lane * KA + (Hn + INn)] = (__bf16)1.0f;  // "one" column
  __syncthreads();

  // ---- per-wave recurrent state ----------------------------------------
  const int b0 = (blockIdx.x * WAVES + wave) * BATCH_PER_WAVE;
  float c_reg[4][8];
  float acc[8];
#pragma unroll
  for (int nt = 0; nt < 4; ++nt)
#pragma unroll
    for (int r = 0; r < 8; ++r) c_reg[nt][r] = 0.0f;
#pragma unroll
  for (int r = 0; r < 8; ++r) acc[r] = 0.0f;

  const float bo_n = (nIdx < SD) ? bo[nIdx] : 0.0f;

  // Wo fragments stay resident across the whole time loop
  const v16bf wo0 = ldB(sWo, lane, 0, Hn);
  const v16bf wo1 = ldB(sWo, lane, 1, Hn);

  // out[b][0][:] = 0
  if (nIdx < SD) {
#pragma unroll
    for (int r = 0; r < 8; ++r) {
      const int m = r + (hi << 3);
      out[(long)(b0 + m) * Ln * SD + nIdx] = 0.0f;
    }
  }

  // ---- time loop: s=0 is the conditioning step (x=0), s>0 are scan steps
  for (int s = 0; s < Ln - 1; ++s) {
    // stage noise + dt columns for this step
    if (lane < 16) {
      const int m = lane;
      const long bm = b0 + m;
      const float4 nz = *(const float4*)(noise + (bm * (Ln - 1) + s) * ND);
      __bf16* rowp = myA + m * KA;
      rowp[Hn + SD + 0] = (__bf16)nz.x;
      rowp[Hn + SD + 1] = (__bf16)nz.y;
      rowp[Hn + SD + 2] = (__bf16)nz.z;
      rowp[Hn + SD + 3] = (__bf16)nz.w;
      const float t0 = t[bm * Ln + s];
      const float t1 = t[bm * Ln + s + 1];
      rowp[Hn + SD + ND] = (__bf16)(t1 - t0);
    }
    lds_fence();

    v16bf a0 = ldA(myA, lane, 0);   // h[0:32]
    v16bf a1 = ldA(myA, lane, 1);   // h[32:64]

    if (s > 0) {
      // x = h @ Wo^T + bo  (only N=0..3 valid); cumsum + feedback into input
      v8f xz = {};
      v8f xf = wmma_bf16(a0, wo0, xz);
      xf = wmma_bf16(a1, wo1, xf);
      if (nIdx < SD) {
#pragma unroll
        for (int r = 0; r < 8; ++r) {
          const int m = r + (hi << 3);
          const float xv = xf[r] + bo_n;
          acc[r] += xv;
          out[(long)(b0 + m) * Ln * SD + (long)s * SD + nIdx] = acc[r];
          myA[m * KA + Hn + nIdx] = (__bf16)xv;
        }
      }
      lds_fence();
    }
    v16bf a2 = ldA(myA, lane, 2);   // [x, noise, dt, 1, pad]

    // gates[16x256] = a[16x96] @ W_aug^T ; 4 hidden chunks x 4 gates x 3 K
#pragma unroll
    for (int nt = 0; nt < 4; ++nt) {
      v8f gi = {}, gf = {}, gg = {}, go = {};
#pragma unroll
      for (int kt = 0; kt < 3; ++kt) {
        const v16bf ak = (kt == 0) ? a0 : ((kt == 1) ? a1 : a2);
        gi = wmma_bf16(ak, ldB(sW + (0 * 4 + nt) * 16 * KA, lane, kt, KA), gi);
        gf = wmma_bf16(ak, ldB(sW + (1 * 4 + nt) * 16 * KA, lane, kt, KA), gf);
        gg = wmma_bf16(ak, ldB(sW + (2 * 4 + nt) * 16 * KA, lane, kt, KA), gg);
        go = wmma_bf16(ak, ldB(sW + (3 * 4 + nt) * 16 * KA, lane, kt, KA), go);
      }
      // lane-local LSTM update (i/f/g/o share lane slots across tiles)
#pragma unroll
      for (int r = 0; r < 8; ++r) {
        const float iv = fast_sigmoid(gi[r]);
        const float fv = fast_sigmoid(gf[r]);
        const float gv = fast_tanh(gg[r]);
        const float ov = fast_sigmoid(go[r]);
        const float c2 = fv * c_reg[nt][r] + iv * gv;
        c_reg[nt][r] = c2;
        const float h2 = ov * fast_tanh(c2);
        const int m = r + (hi << 3);
        myA[m * KA + nt * 16 + nIdx] = (__bf16)h2;   // h back into A buffer
      }
    }
  }

  // ---- final projection x_last and last cumsum row ---------------------
  lds_fence();
  {
    v16bf a0 = ldA(myA, lane, 0);
    v16bf a1 = ldA(myA, lane, 1);
    v8f xz = {};
    v8f xf = wmma_bf16(a0, wo0, xz);
    xf = wmma_bf16(a1, wo1, xf);
    if (nIdx < SD) {
#pragma unroll
      for (int r = 0; r < 8; ++r) {
        const int m = r + (hi << 3);
        acc[r] += xf[r] + bo_n;
        out[(long)(b0 + m) * Ln * SD + (long)(Ln - 1) * SD + nIdx] = acc[r];
      }
    }
  }
}

extern "C" void kernel_launch(void* const* d_in, const int* in_sizes, int n_in,
                              void* d_out, int out_size, void* d_ws, size_t ws_size,
                              hipStream_t stream) {
  const float* noise = (const float*)d_in[0];
  const float* t     = (const float*)d_in[1];
  const float* W_ih  = (const float*)d_in[2];
  const float* W_hh  = (const float*)d_in[3];
  const float* b_ih  = (const float*)d_in[4];
  const float* b_hh  = (const float*)d_in[5];
  const float* Wo    = (const float*)d_in[6];
  const float* bo    = (const float*)d_in[7];
  float* out = (float*)d_out;

  dim3 grid(Bn / BATCH_PER_BLOCK);  // 128 blocks
  dim3 block(TPB);                  // 128 threads = 4 waves
  hipLaunchKernelGGL(genlstm_wmma_kernel, grid, block, 0, stream,
                     noise, t, W_ih, W_hh, b_ih, b_hh, Wo, bo, out);
}
